// Multi_Head_Layer_71966472012351
// MI455X (gfx1250) — compile-verified, matrix operand order checked by reading, not run on hardware
//
#include <hip/hip_runtime.h>
#include <hip/hip_bf16.h>

// ---------------------------------------------------------------------------
// Multi-head attention for MI455X (gfx1250), bf16 WMMA path.
//   B=4, S=2048, D=768, H=8, DH=96.  All matmuls via v_wmma_f32_16x16x32_bf16.
//   DPP16 softmax reductions (base-2, scale pre-folded into Q), double-buffered
//   LDS tiles (1 barrier/iter), TDM descriptors feeding A-tiles and K-tiles.
// ---------------------------------------------------------------------------

typedef __bf16 bf16_t;
typedef __attribute__((ext_vector_type(16))) __bf16 v16bf;
typedef __attribute__((ext_vector_type(8)))  float  v8f;
typedef __attribute__((ext_vector_type(4)))  unsigned v4u;
typedef __attribute__((ext_vector_type(8)))  int      v8i;
typedef __attribute__((ext_vector_type(4)))  int      v4i;

#define BQ     4
#define SEQ    2048
#define DMODEL 768
#define NH     8
#define DHEAD  96
#define MROWS  (BQ * SEQ)   // 8192

__device__ inline unsigned short f2bfu(float f) {   // f32 -> bf16 bits (RNE)
  unsigned u = __builtin_bit_cast(unsigned, f);
  return (unsigned short)((u + 0x7FFFu + ((u >> 16) & 1u)) >> 16);
}
__device__ inline bf16_t f2bf(float f) {
  unsigned short s = f2bfu(f);
  return __builtin_bit_cast(bf16_t, s);
}
__device__ inline v8f zero8() {
  v8f z = {0.f, 0.f, 0.f, 0.f, 0.f, 0.f, 0.f, 0.f};
  return z;
}
__device__ inline v8f wmma_bf16(v16bf a, v16bf b, v8f c) {
  return __builtin_amdgcn_wmma_f32_16x16x32_bf16(false, a, false, b, (short)0, c,
                                                 false, false);
}
__device__ inline float fexp2(float x) {
#if defined(__AMDGCN__) && __has_builtin(__builtin_amdgcn_exp2f)
  return __builtin_amdgcn_exp2f(x);
#else
  return exp2f(x);
#endif
}
__device__ inline float frcp(float x) {
#if defined(__AMDGCN__) && __has_builtin(__builtin_amdgcn_rcpf)
  return __builtin_amdgcn_rcpf(x);
#else
  return 1.f / x;
#endif
}

union FragA {            // 16 bf16 per lane = one A/B operand of 16x16x32 bf16
  v16bf v;
  uint4 q[2];
};

// ---- 16-lane butterfly reductions via DPP16 (fusable into VALU operands) ---
#if defined(__AMDGCN__) && __has_builtin(__builtin_amdgcn_mov_dpp)
template <int CTRL>
__device__ inline float dppf(float x) {
  return __builtin_bit_cast(float,
      __builtin_amdgcn_mov_dpp(__builtin_bit_cast(int, x), CTRL, 0xf, 0xf, true));
}
#elif defined(__AMDGCN__) && __has_builtin(__builtin_amdgcn_update_dpp)
template <int CTRL>
__device__ inline float dppf(float x) {
  int xi = __builtin_bit_cast(int, x);
  return __builtin_bit_cast(float,
      __builtin_amdgcn_update_dpp(xi, xi, CTRL, 0xf, 0xf, false));
}
#endif
#if defined(__AMDGCN__) && (__has_builtin(__builtin_amdgcn_mov_dpp) || \
                            __has_builtin(__builtin_amdgcn_update_dpp))
__device__ inline float redmax16(float x) {
  x = fmaxf(x, dppf<0xB1>(x));    // quad_perm(1,0,3,2)
  x = fmaxf(x, dppf<0x4E>(x));    // quad_perm(2,3,0,1)
  x = fmaxf(x, dppf<0x141>(x));   // row_half_mirror
  x = fmaxf(x, dppf<0x140>(x));   // row_mirror
  return x;
}
__device__ inline float redsum16(float x) {
  x += dppf<0xB1>(x);
  x += dppf<0x4E>(x);
  x += dppf<0x141>(x);
  x += dppf<0x140>(x);
  return x;
}
#else
__device__ inline float redmax16(float x) {
  x = fmaxf(x, __shfl_xor(x, 1, 32));
  x = fmaxf(x, __shfl_xor(x, 2, 32));
  x = fmaxf(x, __shfl_xor(x, 4, 32));
  x = fmaxf(x, __shfl_xor(x, 8, 32));
  return x;
}
__device__ inline float redsum16(float x) {
  x += __shfl_xor(x, 1, 32);
  x += __shfl_xor(x, 2, 32);
  x += __shfl_xor(x, 4, 32);
  x += __shfl_xor(x, 8, 32);
  return x;
}
#endif

// ---- Tensor Data Mover: up to 3D tile global->LDS, padded LDS rows ---------
// LDS layout: rows of tile_d0 elems (64B here) + 16B pad => stride 40 bf16.
#if defined(__AMDGCN__) && __has_builtin(__builtin_amdgcn_tensor_load_to_lds)
#define USE_TDM 1
// data_size=2B | pad_enable | pad every 16 DW | pad 4 DW
#define TDM_PADCFG ((1u << 16) | (1u << 20) | (3u << 22) | (3u << 25))
__device__ inline v8i tdm_make_g1(unsigned t0, unsigned t1, unsigned t2,
                                  unsigned s0, unsigned s1, unsigned g1w0) {
  v8i g1 = {0, 0, 0, 0, 0, 0, 0, 0};
  g1[0] = (int)g1w0;                                  // data_size / pad config
  g1[1] = (int)(t0 << 16);                            // tensor_dim0 lo16
  g1[2] = (int)((t0 >> 16) | (t1 << 16));             // td0 hi | td1 lo
  g1[3] = (int)((t1 >> 16) | (t0 << 16));             // td1 hi | tile_dim0
  g1[4] = (int)(t1 | (t2 << 16));                     // tile_dim1 | tile_dim2
  g1[5] = (int)s0;                                    // tensor_dim0_stride lo32
  g1[6] = (int)((s1 & 0xffffu) << 16);                // dim1_stride lo16
  g1[7] = (int)(s1 >> 16);                            // dim1_stride hi32
  return g1;
}
__device__ inline void tdm_issue(unsigned lds_off, const void* gaddr,
                                 v8i g1, v4i g2) {
  unsigned long long ga = (unsigned long long)(uintptr_t)gaddr;
  v4u g0 = {0, 0, 0, 0};
  g0[0] = 1u;                                         // count = 1 descriptor
  g0[1] = lds_off;                                    // lds_addr (bytes)
  g0[2] = (unsigned)(ga & 0xffffffffu);               // global_addr lo
  g0[3] = (unsigned)((ga >> 32) & 0x1ffffffu) | (2u << 30);  // hi | type=2
  v4i gz = {0, 0, 0, 0};
#if __clang_major__ >= 23
  v8i gz8 = {0, 0, 0, 0, 0, 0, 0, 0};
  __builtin_amdgcn_tensor_load_to_lds(g0, g1, g2, gz, gz8, 0);
#else
  __builtin_amdgcn_tensor_load_to_lds(g0, g1, g2, gz, 0);
#endif
}
#endif

__device__ inline void wait_tensorcnt0() {
#if defined(__AMDGCN__) && __has_builtin(__builtin_amdgcn_s_wait_tensorcnt)
  __builtin_amdgcn_s_wait_tensorcnt(0);
#elif defined(__AMDGCN__)
  asm volatile("s_wait_tensorcnt 0x0" ::: "memory");
#endif
}

// ---------------------------------------------------------------------------
// f32 -> bf16 conversion, vectorized x4 (all sizes divisible by 4)
// ---------------------------------------------------------------------------
__global__ void cvt_f32_bf16(const float* __restrict__ in, bf16_t* __restrict__ out,
                             int n4) {
  int i = blockIdx.x * blockDim.x + threadIdx.x;
  int stride = gridDim.x * blockDim.x;
  for (; i < n4; i += stride) {
    float4 f = ((const float4*)in)[i];
    ushort4 r;
    r.x = f2bfu(f.x); r.y = f2bfu(f.y); r.z = f2bfu(f.z); r.w = f2bfu(f.w);
    ((ushort4*)out)[i] = r;
  }
}

// ---------------------------------------------------------------------------
// Generic bf16 GEMM: C[M,N] = (A[M,K] * B[K,N] + bias[N]) * oscale(QKV only)
//   QKV=true : B/bias indexed per head (blockIdx.z), output bf16 in [B,H,S,DH]
//   QKV=false: output f32 row-major [M,N]
// Tile 128x96x32 (N % 96 == 0), 8 waves = 4x2 of 32x48 (6 wmma / k-step),
// double-buffered LDS, A tiles via TDM.
// ---------------------------------------------------------------------------
template <bool QKV, typename OutT>
__global__ __launch_bounds__(256) void gemm_bf16_k(
    const bf16_t* __restrict__ A, const bf16_t* __restrict__ Bw,
    const float* __restrict__ bias, OutT* __restrict__ C, int M, int N, int K,
    float oscale) {
  constexpr int BM = 128, BN = 96, BK = 32;
  constexpr int ASTR = 40;   // bf16 per As row (32 + 8 pad)  == TDM pad layout
  constexpr int BSTR = 20;   // u32 per Bt row  (16 + 4 pad)
  __shared__ alignas(16) bf16_t   As[2][BM * ASTR];
  __shared__ alignas(16) unsigned Bt32[2][BN * BSTR];

  const int t = threadIdx.x;
  const int w = t >> 5, lane = t & 31;
  const int half = lane >> 4, l16 = lane & 15;
  const int mbase = blockIdx.x * BM;
  const int nbase = blockIdx.y * BN;
  const int h = blockIdx.z;
  const bf16_t* Bh = Bw + (size_t)h * K * N;
  const float* biash = bias + (size_t)h * N;

  const int roff = (w & 3) * 32;     // 4 row groups of 32
  const int coff = (w >> 2) * 48;    // 2 col groups of 48

#ifdef USE_TDM
  const v8i g1A = tdm_make_g1(BK, BM, 0, (unsigned)K, 0, TDM_PADCFG);
  const v4i g2z = {0, 0, 0, 0};
#endif

  auto stageA = [&](int k0, int buf) {
#ifdef USE_TDM
    if (w == 0)
      tdm_issue((unsigned)(uintptr_t)(const void*)&As[buf][0],
                A + (size_t)mbase * K + k0, g1A, g2z);
#else
#pragma unroll
    for (int i = 0; i < 2; ++i) {
      int cc = t + i * 256;           // 512 chunks of 8 bf16
      int row = cc >> 2;
      int ko = (cc & 3) * 8;
      *(uint4*)(&As[buf][row * ASTR + ko]) =
          *(const uint4*)(A + (size_t)(mbase + row) * K + k0 + ko);
    }
#endif
  };
  auto stageB = [&](int k0, int buf) {
    if (t < 192) {                    // 16 k-pairs x 12 col chunks of 8
      int kp = t / 12;
      int n0 = (t - kp * 12) * 8;
      union { uint4 q; unsigned short u[8]; } r0, r1;
      r0.q = *(const uint4*)(Bh + (size_t)(k0 + 2 * kp) * N + nbase + n0);
      r1.q = *(const uint4*)(Bh + (size_t)(k0 + 2 * kp + 1) * N + nbase + n0);
#pragma unroll
      for (int i = 0; i < 8; ++i)
        Bt32[buf][(n0 + i) * BSTR + kp] =
            (unsigned)r0.u[i] | ((unsigned)r1.u[i] << 16);
    }
  };

  v8f acc[2][3];
#pragma unroll
  for (int i = 0; i < 2; ++i)
#pragma unroll
    for (int j = 0; j < 3; ++j) acc[i][j] = zero8();

  stageA(0, 0);
  stageB(0, 0);
  const int nkt = K / BK;
  for (int kt = 0; kt < nkt; ++kt) {
    const int cb = kt & 1, nb = cb ^ 1;
#ifdef USE_TDM
    if (w == 0) wait_tensorcnt0();    // TDM for buffer cb complete
#endif
    __syncthreads();                  // staging for cb visible to all waves
    if (kt + 1 < nkt) {               // overlap next-tile staging with compute
      stageA((kt + 1) * BK, nb);
      stageB((kt + 1) * BK, nb);
    }

    const bf16_t* as = &As[cb][0];
    const unsigned* bt = &Bt32[cb][0];
    FragA a0, a1, b[3];
    a0.q[0] = *(const uint4*)(as + (roff + l16) * ASTR + half * 8);
    a0.q[1] = *(const uint4*)(as + (roff + l16) * ASTR + 16 + half * 8);
    a1.q[0] = *(const uint4*)(as + (roff + 16 + l16) * ASTR + half * 8);
    a1.q[1] = *(const uint4*)(as + (roff + 16 + l16) * ASTR + 16 + half * 8);
#pragma unroll
    for (int j = 0; j < 3; ++j) {
      b[j].q[0] = *(const uint4*)(bt + (coff + j * 16 + l16) * BSTR + half * 4);
      b[j].q[1] = *(const uint4*)(bt + (coff + j * 16 + l16) * BSTR + 8 + half * 4);
      acc[0][j] = wmma_bf16(a0.v, b[j].v, acc[0][j]);
      acc[1][j] = wmma_bf16(a1.v, b[j].v, acc[1][j]);
    }
  }

#pragma unroll
  for (int i = 0; i < 2; ++i) {
#pragma unroll
    for (int j = 0; j < 3; ++j) {
      int n = nbase + coff + j * 16 + l16;
      float bv = biash[n];
#pragma unroll
      for (int v = 0; v < 8; ++v) {
        int r = mbase + roff + i * 16 + half * 8 + v;  // D-layout: M = v + 8*half
        float val = acc[i][j][v] + bv;
        if constexpr (QKV) {
          val *= oscale;              // Q carries softmax scale * log2(e)
          int bb = r / SEQ, s = r - bb * SEQ;
          size_t idx = (((size_t)bb * NH + h) * SEQ + s) * DHEAD + n;
          C[idx] = f2bf(val);
        } else {
          C[(size_t)r * N + n] = val;
        }
      }
    }
  }
}

// ---------------------------------------------------------------------------
// Flash attention per (b,h). Block = 8 waves = 128 query rows; key chunks = 32.
// Q is pre-scaled by (1/sqrt(DH))*log2(e) so softmax runs in pure exp2.
// K tile via one 3D TDM into padded [kc][key][40] LDS; V transposed+packed
// (key,key+16) pairs, key axis permuted consistently for P and V.
// Row-sum kept lane-partial, reduced once at the end. 1 barrier per chunk.
// ---------------------------------------------------------------------------
__global__ __launch_bounds__(256) void attn_k(const bf16_t* __restrict__ Qp,
                                              const bf16_t* __restrict__ Kp,
                                              const bf16_t* __restrict__ Vp,
                                              bf16_t* __restrict__ Cat) {
  constexpr int KROW = 40;   // bf16 per Ks row (32 + 8 pad) == TDM pad layout
  constexpr int PSTR = 20;   // u32 per Vt/Pt row (16 + 4 pad)
  __shared__ alignas(16) bf16_t   Ks[2][3 * 32 * KROW];   // [kc][key][40]
  __shared__ alignas(16) unsigned Vt32[2][DHEAD * PSTR];  // [dh][keypair]
  __shared__ alignas(16) unsigned Pt32[8][16 * PSTR];     // per-wave [row][kp]

  const int t = threadIdx.x;
  const int w = t >> 5, lane = t & 31;
  const int half = lane >> 4, l16 = lane & 15;
  const int bh = blockIdx.y;
  const int b = bh / NH, h = bh - b * NH;
  const size_t base = (size_t)bh * SEQ * DHEAD;
  const int qrow = blockIdx.x * 128 + w * 16;
  const int NC = SEQ / 32;

#ifdef USE_TDM
  const v8i g1K = tdm_make_g1(32, 32, 3, DHEAD, 32, TDM_PADCFG);
  const v4i g2K = {3, 0, 0, 0};
#endif

  auto stageKV = [&](int c, int buf) {
    const int k0 = c * 32;
#ifdef USE_TDM
    if (w == 0)
      tdm_issue((unsigned)(uintptr_t)(const void*)&Ks[buf][0],
                Kp + base + (size_t)k0 * DHEAD, g1K, g2K);
#else
    for (int ck = t; ck < 384; ck += 256) {
      int kc = ck >> 7;
      int rem = ck & 127;
      int key = rem >> 2;
      int ko = (rem & 3) * 8;
      *(uint4*)(&Ks[buf][(kc * 32 + key) * KROW + ko]) =
          *(const uint4*)(Kp + base + (size_t)(k0 + key) * DHEAD + kc * 32 + ko);
    }
#endif
    if (t < 192) {                    // V: transpose + pack (key,key+16) pairs
      int key16 = t / 12;             // 0..15
      int dh0 = (t - key16 * 12) * 8;
      union { uint4 q; unsigned short u[8]; } va, vb;
      va.q = *(const uint4*)(Vp + base + (size_t)(k0 + key16) * DHEAD + dh0);
      vb.q = *(const uint4*)(Vp + base + (size_t)(k0 + key16 + 16) * DHEAD + dh0);
#pragma unroll
      for (int i = 0; i < 8; ++i)
        Vt32[buf][(dh0 + i) * PSTR + key16] =
            (unsigned)va.u[i] | ((unsigned)vb.u[i] << 16);
    }
    if (c + 1 < NC) {                 // GL2 prefetch for the chunk after next
      __builtin_prefetch(Kp + base + (size_t)(k0 + 32 + (t >> 3)) * DHEAD, 0, 1);
      __builtin_prefetch(Vp + base + (size_t)(k0 + 32 + (t >> 3)) * DHEAD, 0, 1);
    }
  };

  // Q fragments (A-layout), rows qrow..+15, k = dh in 3 chunks of 32
  FragA qa[3];
#pragma unroll
  for (int kc = 0; kc < 3; ++kc) {
    const bf16_t* qp = Qp + base + (size_t)(qrow + l16) * DHEAD + kc * 32 + half * 8;
    qa[kc].q[0] = *(const uint4*)qp;
    qa[kc].q[1] = *(const uint4*)(qp + 16);
  }

  v8f o[6];
#pragma unroll
  for (int n = 0; n < 6; ++n) o[n] = zero8();
  float m_i[8], l_i[8];
#pragma unroll
  for (int v = 0; v < 8; ++v) { m_i[v] = -1e30f; l_i[v] = 0.f; }

  stageKV(0, 0);
  for (int c = 0; c < NC; ++c) {
    const int cb = c & 1, nb = cb ^ 1;
#ifdef USE_TDM
    if (w == 0) wait_tensorcnt0();    // K tile for buffer cb complete
#endif
    __syncthreads();
    if (c + 1 < NC) stageKV(c + 1, nb);   // overlap staging with compute

    // ---- scores: two 16x16 tiles (keys 0-15, 16-31) over dh in 3 wmma each --
    v8f s0 = zero8(), s1 = zero8();
#pragma unroll
    for (int kc = 0; kc < 3; ++kc) {
      FragA kb0, kb1;
      const bf16_t* kp0 = &Ks[cb][(kc * 32 + l16) * KROW + half * 8];
      kb0.q[0] = *(const uint4*)kp0;
      kb0.q[1] = *(const uint4*)(kp0 + 16);
      const bf16_t* kp1 = &Ks[cb][(kc * 32 + 16 + l16) * KROW + half * 8];
      kb1.q[0] = *(const uint4*)kp1;
      kb1.q[1] = *(const uint4*)(kp1 + 16);
      s0 = wmma_bf16(qa[kc].v, kb0.v, s0);
      s1 = wmma_bf16(qa[kc].v, kb1.v, s1);
    }

    // ---- online softmax (base 2): lane owns rows M = v + 8*half ------------
    float alpha[8];
#pragma unroll
    for (int v = 0; v < 8; ++v) {
      float x0 = s0[v];
      float x1 = s1[v];
      float cm = redmax16(fmaxf(x0, x1));
      float mn = fmaxf(m_i[v], cm);
      alpha[v] = fexp2(m_i[v] - mn);
      float p0 = fexp2(x0 - mn);
      float p1 = fexp2(x1 - mn);
      l_i[v] = l_i[v] * alpha[v] + (p0 + p1);   // lane-partial row sum
      m_i[v] = mn;
      int r = v + 8 * half;
      Pt32[w][r * PSTR + l16] =        // packed (key, key+16) pair
          (unsigned)f2bfu(p0) | ((unsigned)f2bfu(p1) << 16);
    }
#pragma unroll
    for (int n = 0; n < 6; ++n)
#pragma unroll
      for (int v = 0; v < 8; ++v) o[n][v] *= alpha[v];

    // wave-private LDS round-trip: P stores -> fragment load
    asm volatile("s_wait_dscnt 0x0" ::: "memory");

    FragA pa;
    pa.q[0] = *(const uint4*)(&Pt32[w][l16 * PSTR + half * 4]);
    pa.q[1] = *(const uint4*)(&Pt32[w][l16 * PSTR + 8 + half * 4]);

    // ---- O[16x96] += P[16x32] * V[32x96] (key axis consistently permuted) ---
#pragma unroll
    for (int n = 0; n < 6; ++n) {
      FragA vb;
      vb.q[0] = *(const uint4*)(&Vt32[cb][(n * 16 + l16) * PSTR + half * 4]);
      vb.q[1] = *(const uint4*)(&Vt32[cb][(n * 16 + l16) * PSTR + 8 + half * 4]);
      o[n] = wmma_bf16(pa.v, vb.v, o[n]);
    }
  }

  // ---- final row-sum reduction, normalize, write concat [B,S,H*DH] bf16 ----
#pragma unroll
  for (int v = 0; v < 8; ++v) {
    float inv = frcp(redsum16(l_i[v]));
    int s = qrow + v + 8 * half;
    size_t rowb = ((size_t)b * SEQ + s) * (NH * DHEAD) + (size_t)h * DHEAD;
#pragma unroll
    for (int n = 0; n < 6; ++n) Cat[rowb + n * 16 + l16] = f2bf(o[n][v] * inv);
  }
}

// ---------------------------------------------------------------------------
// Host launcher
// ---------------------------------------------------------------------------
extern "C" void kernel_launch(void* const* d_in, const int* in_sizes, int n_in,
                              void* d_out, int out_size, void* d_ws, size_t ws_size,
                              hipStream_t stream) {
  (void)in_sizes; (void)n_in; (void)out_size; (void)ws_size;

  const float* Xq = (const float*)d_in[0];
  const float* Xk = (const float*)d_in[1];
  const float* Xv = (const float*)d_in[2];
  const float* Wq = (const float*)d_in[3];
  const float* bq = (const float*)d_in[4];
  const float* Wk = (const float*)d_in[5];
  const float* bk = (const float*)d_in[6];
  const float* Wv = (const float*)d_in[7];
  const float* bv = (const float*)d_in[8];
  const float* Wo = (const float*)d_in[9];
  const float* bo = (const float*)d_in[10];
  float* out = (float*)d_out;

  const size_t nX = (size_t)BQ * SEQ * DMODEL;     // 6,291,456
  const size_t nW = (size_t)NH * DMODEL * DHEAD;   //   589,824
  const size_t nQ = (size_t)BQ * NH * SEQ * DHEAD; // 6,291,456
  bf16_t* ws  = (bf16_t*)d_ws;                     // ~93 MB bf16 workspace
  bf16_t* pXq = ws;          bf16_t* pXk = pXq + nX;  bf16_t* pXv = pXk + nX;
  bf16_t* pWq = pXv + nX;    bf16_t* pWk = pWq + nW;  bf16_t* pWv = pWk + nW;
  bf16_t* pWo = pWv + nW;
  bf16_t* pQ  = pWo + nW;    bf16_t* pK  = pQ + nQ;   bf16_t* pV  = pK + nQ;
  bf16_t* pCat = pV + nQ;

  // 1) precision demotion f32 -> bf16 (RNE), vectorized x4
  cvt_f32_bf16<<<1024, 256, 0, stream>>>(Xq, pXq, (int)(nX / 4));
  cvt_f32_bf16<<<1024, 256, 0, stream>>>(Xk, pXk, (int)(nX / 4));
  cvt_f32_bf16<<<1024, 256, 0, stream>>>(Xv, pXv, (int)(nX / 4));
  cvt_f32_bf16<<<256, 256, 0, stream>>>(Wq, pWq, (int)(nW / 4));
  cvt_f32_bf16<<<256, 256, 0, stream>>>(Wk, pWk, (int)(nW / 4));
  cvt_f32_bf16<<<256, 256, 0, stream>>>(Wv, pWv, (int)(nW / 4));
  cvt_f32_bf16<<<256, 256, 0, stream>>>(Wo, pWo, (int)(nW / 4));

  // softmax scale * log2(e), folded into the Q projection output
  const float qscale = 0.10206207261596575f * 1.4426950408889634f;

  // 2) per-head QKV projections: [8192,768] x [768,96] + bias -> [B,H,S,DH]
  dim3 gqkv(MROWS / 128, DHEAD / 96, NH);   // N = 96 -> one column block
  gemm_bf16_k<true, bf16_t><<<gqkv, 256, 0, stream>>>(pXq, pWq, bq, pQ,
                                                      MROWS, DHEAD, DMODEL, qscale);
  gemm_bf16_k<true, bf16_t><<<gqkv, 256, 0, stream>>>(pXk, pWk, bk, pK,
                                                      MROWS, DHEAD, DMODEL, 1.0f);
  gemm_bf16_k<true, bf16_t><<<gqkv, 256, 0, stream>>>(pXv, pWv, bv, pV,
                                                      MROWS, DHEAD, DMODEL, 1.0f);

  // 3) attention -> concat [B,S,H*DH] bf16
  attn_k<<<dim3(SEQ / 128, BQ * NH), 256, 0, stream>>>(pQ, pK, pV, pCat);

  // 4) output projection: [8192,768] x [768,768] + bo -> f32 d_out
  gemm_bf16_k<false, float><<<dim3(MROWS / 128, DMODEL / 96, 1), 256, 0, stream>>>(
      pCat, pWo, bo, out, MROWS, DMODEL, DMODEL, 1.0f);
}
